// Program_26482768347954
// MI455X (gfx1250) — compile-verified, matrix operand order checked by reading, not run on hardware
//
#include <hip/hip_runtime.h>
#include <hip/hip_bf16.h>

typedef __attribute__((ext_vector_type(16))) _Float16 v16h;
typedef __attribute__((ext_vector_type(8)))  float    v8f;

#define LL 64
#define STEPS 10
#define WPB 8   // waves per block (256 threads, wave32)

// K index covered by half-slot h (0..15) of a v16h A/B operand,
// lanes 0-15 (hi=0) vs lanes 16-31 (hi=1), per CDNA5 ISA 7.12.2:
//  lo lanes: VGPR0..3 -> K=0..7, VGPR4..7 -> K=16..23
//  hi lanes: VGPR0..3 -> K=8..15, VGPR4..7 -> K=24..31
__device__ __forceinline__ int kmap(int h, int hi) {
  return (h < 8) ? (h + (hi ? 8 : 0)) : (h + (hi ? 16 : 8));
}

__global__ __launch_bounds__(256, 1) void mlp_scan_kernel(
    const float* __restrict__ x,
    const float* __restrict__ W1, const float* __restrict__ b1,
    const float* __restrict__ W2, const float* __restrict__ b2,
    const float* __restrict__ W3, const float* __restrict__ b3,
    float* __restrict__ out, int nrows)
{
  __shared__ float4 sW1pk[LL];   // (W1[0][k], W1[1][k], W1[2][k], b1[k])

  const int tid  = threadIdx.x;
  const int lane = tid & 31;
  const int wave = tid >> 5;
  const int col  = lane & 15;
  const int hi   = lane >> 4;

  if (tid < LL) {
    sW1pk[tid] = make_float4(W1[tid], W1[LL + tid], W1[2*LL + tid], b1[tid]);
  }
  __syncthreads();

  // W2 (64x64, row-major [k][n]) resident in registers as WMMA-B operands:
  // 4 N-tiles x 2 K-chunks, f16. Loaded once, reused for all 10 steps.
  v16h Bt[4][2];
  #pragma unroll
  for (int t = 0; t < 4; ++t) {
    #pragma unroll
    for (int kr = 0; kr < 2; ++kr) {
      v16h tmp;
      #pragma unroll
      for (int h = 0; h < 16; ++h) {
        int k = kr * 32 + kmap(h, hi);
        tmp[h] = (_Float16)W2[k * LL + t * 16 + col];
      }
      Bt[t][kr] = tmp;
    }
  }

  float b2v[4], w3v[4];
  #pragma unroll
  for (int t = 0; t < 4; ++t) {
    b2v[t] = b2[t * 16 + col];
    w3v[t] = W3[t * 16 + col];
  }
  const float b3s = b3[0];

  const long gw = (long)blockIdx.x * WPB + wave;
  const int  row  = (int)(gw * 16) + col;          // this lane's row (dup across halves)
  const int  rowc = row < nrows ? row : nrows - 1; // clamp: keep EXEC all-1 for WMMA
  const float* xr = x + (size_t)rowc * 7;
  float x0 = xr[0], x1 = xr[1], x2 = xr[2], x6 = xr[6];

  const float LATC  = 5.0f * 0.03f / 200.0f + 0.06f;
  const float GRADC = 1.0f / 0.06f;
  const float SENDC = 2.0f / 3.0f;

  float lx3 = 0.0f, lx4 = 0.0f, lx5 = 0.0f;

  for (int s = 0; s < STEPS; ++s) {
    float g  = (x6 <= 0.0f) ? x2 * (x6 + 1.0f) : x2 * (1.0f - x6);
    float f4 = g + LATC;
    float f5 = g * SENDC;
    float f3 = f4 * GRADC;
    lx3 = f3; lx4 = f4; lx5 = f5;

    // Layer 1 (K=3) on VALU, emitted directly in WMMA-A layout (no transpose).
    v16h A0, A1;
    #pragma unroll
    for (int h = 0; h < 16; ++h) {
      int k0 = kmap(h, hi);
      float4 p = sW1pk[k0];
      float v = fmaf(f3, p.x, fmaf(f4, p.y, fmaf(f5, p.z, p.w)));
      A0[h] = (_Float16)fmaxf(v, 0.0f);
      float4 q = sW1pk[32 + k0];
      float w = fmaf(f3, q.x, fmaf(f4, q.y, fmaf(f5, q.z, q.w)));
      A1[h] = (_Float16)fmaxf(w, 0.0f);
    }

    // Layer 2: h1(16x64) @ W2(64x64) -> 8x v_wmma_f32_16x16x32_f16
    float srow[8];
    #pragma unroll
    for (int r = 0; r < 8; ++r) srow[r] = 0.0f;

    #pragma unroll
    for (int t = 0; t < 4; ++t) {
      v8f acc = {};
      acc = __builtin_amdgcn_wmma_f32_16x16x32_f16(false, A0, false, Bt[t][0],
                                                   (short)0, acc, false, false);
      acc = __builtin_amdgcn_wmma_f32_16x16x32_f16(false, A1, false, Bt[t][1],
                                                   (short)0, acc, false, false);
      // Layer 3 partials: C-layout reg r holds rows r (lo lanes) / r+8 (hi lanes),
      // column n = 16*t + col. Fold in b2, ReLU, scale by W3[n].
      #pragma unroll
      for (int r = 0; r < 8; ++r) {
        float h2 = fmaxf(acc[r] + b2v[t], 0.0f);
        srow[r] = fmaf(h2, w3v[t], srow[r]);
      }
    }

    // Reduce over the 16 lanes of each half-wave (columns), then sigmoid.
    #pragma unroll
    for (int r = 0; r < 8; ++r) {
      float v = srow[r];
      v += __shfl_xor(v, 1, 32);
      v += __shfl_xor(v, 2, 32);
      v += __shfl_xor(v, 4, 32);
      v += __shfl_xor(v, 8, 32);
      srow[r] = 1.0f / (1.0f + __expf(-(v + b3s)));
    }

    // Route row m's sigmoid back to lane m (both halves) as next x2.
    // Lane l selects srow[l&7] -> holds row (l&7)+(l>=16?8:0); then a
    // cross-half shuffle fetches row (lane&15).
    float v = srow[0];
    #pragma unroll
    for (int r = 1; r < 8; ++r) v = ((lane & 7) == r) ? srow[r] : v;
    x2 = __shfl(v, (lane & 7) + ((lane & 8) ? 16 : 0), 32);
  }

  if (hi == 0 && row < nrows) {
    float* o = out + (size_t)row * 7;
    o[0] = x0 + (float)STEPS;
    o[1] = x1;
    o[2] = x2;
    o[3] = lx3;
    o[4] = lx4;
    o[5] = lx5;
    o[6] = x6;
  }
}

extern "C" void kernel_launch(void* const* d_in, const int* in_sizes, int n_in,
                              void* d_out, int out_size, void* d_ws, size_t ws_size,
                              hipStream_t stream) {
  const float* x  = (const float*)d_in[0];
  const float* W1 = (const float*)d_in[1];
  const float* b1 = (const float*)d_in[2];
  const float* W2 = (const float*)d_in[3];
  const float* b2 = (const float*)d_in[4];
  const float* W3 = (const float*)d_in[5];
  const float* b3 = (const float*)d_in[6];
  const int nrows   = in_sizes[0] / 7;
  const int nwaves  = (nrows + 15) / 16;
  const int nblocks = (nwaves + WPB - 1) / WPB;
  mlp_scan_kernel<<<nblocks, 32 * WPB, 0, stream>>>(
      x, W1, b1, W2, b2, W3, b3, (float*)d_out, nrows);
}
